// CausalAttNet_9242769621980
// MI455X (gfx1250) — matrix operand order repaired and checked by reading, not compiled, not run on hardware
//
#include <hip/hip_runtime.h>
#include <hip/hip_bf16.h>

// ---------------- problem constants ----------------
#define Bg   128
#define NPG  512
#define EPG  2048
#define Nn   (Bg*NPG)      // 65536 nodes
#define Ee   (Bg*EPG)      // 262144 edges
#define FIN  768
#define Cc   128
#define Kk   1024          // top-K per group
#define EK   (Bg*Kk)       // 131072 kept edges
#define DK   (Ee-EK)       // 131072 dropped edges

typedef __attribute__((ext_vector_type(16))) __bf16 v16bf;
typedef __attribute__((ext_vector_type(8)))  float  v8f;

union FragBF {
  v16bf v;
  uint4 q[2];
};

__device__ __forceinline__ unsigned short f2bf(float f) {
  unsigned int u = __float_as_uint(f);
  u += 0x7fffu + ((u >> 16) & 1u);   // round-to-nearest-even
  return (unsigned short)(u >> 16);
}

// ---------------- graph normalization ----------------
__global__ void k_deg(const int* __restrict__ ei, const float* __restrict__ w,
                      float* __restrict__ deg) {
  int e = blockIdx.x * blockDim.x + threadIdx.x;
  if (e < Ee) atomicAdd(&deg[ei[Ee + e]], w[e]);
}

__global__ void k_dinv(const float* __restrict__ deg, float* __restrict__ dinv) {
  int i = blockIdx.x * blockDim.x + threadIdx.x;
  if (i < Nn) {
    float d = deg[i];
    dinv[i] = (d > 0.f) ? rsqrtf(fmaxf(d, 1e-12f)) : 0.f;
  }
}

__global__ void k_norm(const int* __restrict__ ei, const float* __restrict__ w,
                       const float* __restrict__ dinv, float* __restrict__ nrm) {
  int e = blockIdx.x * blockDim.x + threadIdx.x;
  if (e < Ee) nrm[e] = dinv[ei[e]] * w[e] * dinv[ei[Ee + e]];
}

// ---------------- f32 -> bf16 conversion (vec4) ----------------
__global__ void k_cvt_bf16(const float* __restrict__ src, unsigned short* __restrict__ dst, int n4) {
  int t = blockIdx.x * blockDim.x + threadIdx.x;
  if (t < n4) {
    float4 v = ((const float4*)src)[t];
    ushort4 o;
    o.x = f2bf(v.x); o.y = f2bf(v.y); o.z = f2bf(v.z); o.w = f2bf(v.w);
    ((ushort4*)dst)[t] = o;
  }
}

// ---------------- pack weight (Kd x Nc, row-major f32) into WMMA B-fragment layout ----------------
// packed index p = ((kt*(Nc/16)+nt)*32 + lane)*16 + e ; lane holds col = nt*16+(lane&15),
// k = kt*32 + (lane>>4)*8 + (e&7) + (e>>3)*16  (16-bit B fragment: lanes 0-15 K 0..7,16..23, lanes 16-31 K 8..15,24..31)
__global__ void k_pack_w(const float* __restrict__ W, unsigned short* __restrict__ pk,
                         int Kd, int Nc) {
  int p = blockIdx.x * blockDim.x + threadIdx.x;
  int total = Kd * Nc;
  if (p >= total) return;
  int e    = p & 15;
  int lane = (p >> 4) & 31;
  int tile = p >> 9;
  int ntn  = Nc >> 4;
  int kt   = tile / ntn;
  int nt   = tile - kt * ntn;
  int col  = (nt << 4) + (lane & 15);
  int k    = (kt << 5) + ((lane >> 4) << 3) + (e & 7) + ((e >> 3) << 4);
  pk[p] = f2bf(W[(size_t)k * Nc + col]);
}

// ---------------- WMMA bf16 GEMM: C[M x Nc] = A[M x Kd] @ W ----------------
// block = 256 threads = 8 waves; each wave: 16 rows x Nc (NT = Nc/16 tiles)
template<int NT>
__global__ __launch_bounds__(256, 1) void k_gemm_bf16(
    const unsigned short* __restrict__ A,
    const unsigned short* __restrict__ Wpk,
    float* __restrict__ C, int Kd, int Nc) {
  const int lane = threadIdx.x & 31;
  const int wave = threadIdx.x >> 5;
  const int rowbase = blockIdx.x * 128 + wave * 16;
  const int row = rowbase + (lane & 15);
  const int kba = (lane >> 4) << 3;          // A frag K sub-offset: 0 or 8
  v8f acc[NT];
#pragma unroll
  for (int i = 0; i < NT; ++i)
#pragma unroll
    for (int r = 0; r < 8; ++r) acc[i][r] = 0.f;

  const unsigned short* arow = A + (size_t)row * Kd;
  const int ksteps = Kd >> 5;
  for (int kt = 0; kt < ksteps; ++kt) {
    const int k0 = kt << 5;
    if (kt + 1 < ksteps) __builtin_prefetch(arow + k0 + 32, 0, 1);
    FragBF a;
    a.q[0] = *(const uint4*)(arow + k0 + kba);
    a.q[1] = *(const uint4*)(arow + k0 + kba + 16);
#pragma unroll
    for (int nt = 0; nt < NT; ++nt) {
      FragBF b;
      const unsigned short* bp = Wpk + (((size_t)(kt * NT + nt)) << 9) + (lane << 4);
      b.q[0] = *(const uint4*)(bp);
      b.q[1] = *(const uint4*)(bp + 8);
      acc[nt] = __builtin_amdgcn_wmma_f32_16x16x32_bf16(
          false, a.v, false, b.v, (short)0, acc[nt], false, false);
    }
  }
  const int mhi = (lane >> 4) << 3;
#pragma unroll
  for (int nt = 0; nt < NT; ++nt) {
    const int n = (nt << 4) + (lane & 15);
#pragma unroll
    for (int r = 0; r < 8; ++r)
      C[(size_t)(rowbase + mhi + r) * Nc + n] = acc[nt][r];
  }
}

// ---------------- edge scatter: agg[col] += Hi[row] * norm ----------------
__global__ void k_scatter(const int* __restrict__ ei, const float* __restrict__ nrm,
                          const float* __restrict__ H, float* __restrict__ agg) {
  int t = blockIdx.x * blockDim.x + threadIdx.x;   // Ee*32 threads
  int e = t >> 5;
  int c = (t & 31) << 2;
  int r  = ei[e];
  int cl = ei[Ee + e];
  float nm = nrm[e];
  float4 v = *(const float4*)(H + (size_t)r * Cc + c);
  float* dst = agg + (size_t)cl * Cc + c;
  atomicAdd(dst + 0, v.x * nm);
  atomicAdd(dst + 1, v.y * nm);
  atomicAdd(dst + 2, v.z * nm);
  atomicAdd(dst + 3, v.w * nm);
}

// ---------------- combine: h = relu(agg + Hr + b); write bf16 (+ optional f32) ----------------
__global__ void k_combine(const float* __restrict__ agg, const float* __restrict__ Hr,
                          const float* __restrict__ bias,
                          unsigned short* __restrict__ hbf, float* __restrict__ hf) {
  int t = blockIdx.x * blockDim.x + threadIdx.x;   // Nn*32 threads
  int i = t >> 5;
  int c = (t & 31) << 2;
  float4 a = *(const float4*)(agg + (size_t)i * Cc + c);
  float4 b = *(const float4*)(Hr  + (size_t)i * Cc + c);
  float v0 = fmaxf(a.x + b.x + bias[c + 0], 0.f);
  float v1 = fmaxf(a.y + b.y + bias[c + 1], 0.f);
  float v2 = fmaxf(a.z + b.z + bias[c + 2], 0.f);
  float v3 = fmaxf(a.w + b.w + bias[c + 3], 0.f);
  ushort4 o; o.x = f2bf(v0); o.y = f2bf(v1); o.z = f2bf(v2); o.w = f2bf(v3);
  *(ushort4*)(hbf + (size_t)i * Cc + c) = o;
  if (hf) {
    float4 f; f.x = v0; f.y = v1; f.z = v2; f.w = v3;
    *(float4*)(hf + (size_t)i * Cc + c) = f;
  }
}

// ---------------- fused edge MLP: score = relu(concat(h[row],h[col])@Wm1+bm1)@Wm2 + bm2 ----
// block = 128 threads = 4 waves; 64 edges per block. The 64x256 bf16 edge-rep tile is
// gathered straight into LDS (no HBM intermediate). Row pitch 264 bf16 = 132 dwords
// (== 4 mod 64 banks) -> the 16-row A-fragment ds_load_b128s are bank-conflict free.
#define REP_PITCH 264
#define MLP_ROWS  64

__global__ __launch_bounds__(128, 1) void k_mlp_score(
    const int* __restrict__ ei, const unsigned short* __restrict__ h,
    const unsigned short* __restrict__ Wpk,
    const float* __restrict__ bm1, const float* __restrict__ Wm2,
    const float* __restrict__ bm2, float* __restrict__ score) {
  __shared__ unsigned short srep[MLP_ROWS * REP_PITCH];   // 33792 B
  const int tid   = threadIdx.x;
  const int ebase = blockIdx.x * MLP_ROWS;

  // cooperative gather: 64 edges * 32 chunks of 8 bf16 = 2048 uint4
#pragma unroll
  for (int i = 0; i < 16; ++i) {
    int idx = tid + i * 128;           // 0..2047
    int el  = idx >> 5;                // local edge
    int q   = idx & 31;                // chunk within row
    int node = (q < 16) ? ei[ebase + el] : ei[Ee + ebase + el];
    uint4 v = *(const uint4*)(h + (size_t)node * Cc + (q & 15) * 8);
    *(uint4*)(&srep[el * REP_PITCH + q * 8]) = v;
  }
  __syncthreads();

  const int lane = tid & 31;
  const int wave = tid >> 5;
  const int rloc = wave * 16 + (lane & 15);
  const int kba  = (lane >> 4) << 3;
  const unsigned short* arow = &srep[rloc * REP_PITCH];

  float sacc[8];
#pragma unroll
  for (int r = 0; r < 8; ++r) sacc[r] = 0.f;

  for (int nc = 0; nc < 8; ++nc) {          // 8 N-chunks of 64 -> acc[4] keeps VGPRs low
    v8f acc[4];
#pragma unroll
    for (int i = 0; i < 4; ++i)
#pragma unroll
      for (int r = 0; r < 8; ++r) acc[i][r] = 0.f;

#pragma unroll
    for (int kt = 0; kt < 8; ++kt) {
      FragBF a;
      a.q[0] = *(const uint4*)(arow + (kt << 5) + kba);
      a.q[1] = *(const uint4*)(arow + (kt << 5) + kba + 16);
#pragma unroll
      for (int nt = 0; nt < 4; ++nt) {
        const int tile = kt * 32 + nc * 4 + nt;   // Wm1 has 512/16 = 32 N-tiles
        FragBF b;
        const unsigned short* bp = Wpk + ((size_t)tile << 9) + (lane << 4);
        b.q[0] = *(const uint4*)(bp);
        b.q[1] = *(const uint4*)(bp + 8);
        acc[nt] = __builtin_amdgcn_wmma_f32_16x16x32_bf16(
            false, a.v, false, b.v, (short)0, acc[nt], false, false);
      }
    }
    // fused layer-2 epilogue: relu + bias, dot with Wm2
#pragma unroll
    for (int nt = 0; nt < 4; ++nt) {
      const int n = (nc << 6) + (nt << 4) + (lane & 15);
      const float wn = Wm2[n];
      const float bn = bm1[n];
#pragma unroll
      for (int r = 0; r < 8; ++r)
        sacc[r] += fmaxf(acc[nt][r] + bn, 0.f) * wn;
    }
  }
  // reduce over the 16 lanes that share each output row (xor 1,2,4,8 stays in half-wave)
#pragma unroll
  for (int r = 0; r < 8; ++r) {
    float s = sacc[r];
    s += __shfl_xor(s, 1, 32);
    s += __shfl_xor(s, 2, 32);
    s += __shfl_xor(s, 4, 32);
    s += __shfl_xor(s, 8, 32);
    sacc[r] = s;
  }
  if ((lane & 15) == 0) {
    const int mhi = (lane >> 4) << 3;
#pragma unroll
    for (int r = 0; r < 8; ++r)
      score[ebase + wave * 16 + mhi + r] = sacc[r] + bm2[0];
  }
}

// ---------------- per-group bitonic sort (descending by score, ties by index asc) ----------------
__global__ __launch_bounds__(512) void k_sort_group(const float* __restrict__ score,
                                                    int* __restrict__ order) {
  __shared__ float sk[EPG];
  __shared__ int   si[EPG];
  const int g = blockIdx.x;
  for (int i = threadIdx.x; i < EPG; i += blockDim.x) {
    sk[i] = score[g * EPG + i];
    si[i] = i;
  }
  __syncthreads();
  for (int k = 2; k <= EPG; k <<= 1) {
    for (int j = k >> 1; j > 0; j >>= 1) {
      for (int i = threadIdx.x; i < EPG; i += blockDim.x) {
        int ixj = i ^ j;
        if (ixj > i) {
          float sa = sk[i], sb = sk[ixj];
          int   ia = si[i], ib = si[ixj];
          bool before = (sa > sb) || (sa == sb && ia < ib);   // i belongs before ixj (desc)
          bool up = ((i & k) == 0);
          if (up != before) {
            sk[i] = sb; sk[ixj] = sa;
            si[i] = ib; si[ixj] = ia;
          }
        }
      }
      __syncthreads();
    }
  }
  for (int i = threadIdx.x; i < EPG; i += blockDim.x) order[g * EPG + i] = si[i];
}

// ---------------- mark nodes touched by selected edges ----------------
__global__ void k_mark(const int* __restrict__ ei, const int* __restrict__ order,
                       int* __restrict__ mask, int offset, int cnt) {
  int s = blockIdx.x * blockDim.x + threadIdx.x;
  if (s >= cnt) return;
  int b = s / Kk;
  int j = s - b * Kk;
  int ge = b * EPG + order[b * EPG + offset + j];
  mask[ei[ge]] = 1;
  mask[ei[Ee + ge]] = 1;
}

// ---------------- two-level inclusive prefix scan over N=65536 ints ----------------
__global__ void k_scan1(const int* __restrict__ mask, int* __restrict__ cum,
                        int* __restrict__ bsum) {
  __shared__ int sh[256];
  int i = blockIdx.x * 256 + threadIdx.x;
  int v = mask[i];
  sh[threadIdx.x] = v;
  __syncthreads();
  for (int off = 1; off < 256; off <<= 1) {
    int t = (threadIdx.x >= off) ? sh[threadIdx.x - off] : 0;
    __syncthreads();
    sh[threadIdx.x] += t;
    __syncthreads();
  }
  cum[i] = sh[threadIdx.x];
  if (threadIdx.x == 255) bsum[blockIdx.x] = sh[255];
}

__global__ void k_scan2(const int* __restrict__ bsum, int* __restrict__ boff) {
  __shared__ int sh[256];
  int v = bsum[threadIdx.x];
  sh[threadIdx.x] = v;
  __syncthreads();
  for (int off = 1; off < 256; off <<= 1) {
    int t = (threadIdx.x >= off) ? sh[threadIdx.x - off] : 0;
    __syncthreads();
    sh[threadIdx.x] += t;
    __syncthreads();
  }
  boff[threadIdx.x] = sh[threadIdx.x] - v;   // exclusive
}

__global__ void k_scan3(int* __restrict__ cum, const int* __restrict__ boff) {
  int i = blockIdx.x * 256 + threadIdx.x;
  cum[i] += boff[blockIdx.x];
}

// ---------------- selected-edge outputs (ei remapped, attr, w) ----------------
__global__ void k_edge_out(const int* __restrict__ ei, const int* __restrict__ order,
                           const int* __restrict__ cum, const float* __restrict__ attr,
                           const float* __restrict__ score,
                           float* __restrict__ oEi, float* __restrict__ oAttr,
                           float* __restrict__ oW, int offset, int cnt, float wsign) {
  int s = blockIdx.x * blockDim.x + threadIdx.x;
  if (s >= cnt) return;
  int b = s / Kk;
  int j = s - b * Kk;
  int ge = b * EPG + order[b * EPG + offset + j];
  oEi[s]       = (float)(cum[ei[ge]] - 1);
  oEi[cnt + s] = (float)(cum[ei[Ee + ge]] - 1);
  oAttr[s] = attr[ge];
  oW[s]    = wsign * score[ge];
}

// ---------------- permuted node features (stable: masked first, then unmasked) ----------------
__global__ void k_node_x(const int* __restrict__ mask, const int* __restrict__ cum,
                         const float* __restrict__ h, float* __restrict__ ox) {
  int t = blockIdx.x * blockDim.x + threadIdx.x;   // Nn*32 threads
  int i = t >> 5;
  int c = (t & 31) << 2;
  int ci = cum[i];
  int total = cum[Nn - 1];
  int p = mask[i] ? (ci - 1) : (total + i - ci);
  float4 v = *(const float4*)(h + (size_t)i * Cc + c);
  float* d = ox + (size_t)p * Cc + c;               // dest region may be 4B-misaligned
  d[0] = v.x; d[1] = v.y; d[2] = v.z; d[3] = v.w;
}

__global__ void k_node_meta(const int* __restrict__ mask, const int* __restrict__ cum,
                            const int* __restrict__ batch, float* __restrict__ obatch,
                            float* __restrict__ on) {
  int i = blockIdx.x * blockDim.x + threadIdx.x;
  if (i >= Nn) return;
  int ci = cum[i];
  int total = cum[Nn - 1];
  int p = mask[i] ? (ci - 1) : (total + i - ci);
  obatch[p] = (float)batch[i];
  if (i == 0) on[0] = (float)total;
}

__global__ void k_copy(const float* __restrict__ s, float* __restrict__ d, int n) {
  int i = blockIdx.x * blockDim.x + threadIdx.x;
  if (i < n) d[i] = s[i];
}

// ================= host orchestration =================
static inline size_t alignUp(size_t x) { return (x + 255) & ~(size_t)255; }

extern "C" void kernel_launch(void* const* d_in, const int* in_sizes, int n_in,
                              void* d_out, int out_size, void* d_ws, size_t ws_size,
                              hipStream_t stream) {
  (void)in_sizes; (void)n_in; (void)out_size; (void)ws_size;
  const float* x    = (const float*)d_in[0];
  const int*   ei   = (const int*)  d_in[1];
  const float* attr = (const float*)d_in[2];
  const int*   bat  = (const int*)  d_in[3];
  const float* W1i  = (const float*)d_in[4];
  const float* W1r  = (const float*)d_in[5];
  const float* b1   = (const float*)d_in[6];
  const float* W2i  = (const float*)d_in[7];
  const float* W2r  = (const float*)d_in[8];
  const float* b2   = (const float*)d_in[9];
  const float* Wm1  = (const float*)d_in[10];
  const float* bm1  = (const float*)d_in[11];
  const float* Wm2  = (const float*)d_in[12];
  const float* bm2  = (const float*)d_in[13];

  // ---- workspace layout (bump allocator, 256B aligned) ----
  char* ws = (char*)d_ws;
  size_t off = 0;
  auto take = [&](size_t bytes) { size_t o = off; off = alignUp(off + bytes); return o; };
  size_t oDeg  = take((size_t)Nn * 4);
  size_t oDinv = take((size_t)Nn * 4);
  size_t oNorm = take((size_t)Ee * 4);
  size_t oXbf  = take((size_t)Nn * FIN * 2);
  size_t oHi   = take((size_t)Nn * Cc * 4);
  size_t oHr   = take((size_t)Nn * Cc * 4);
  size_t oAgg  = take((size_t)Nn * Cc * 4);
  size_t oH1bf = take((size_t)Nn * Cc * 2);
  size_t oH2f  = take((size_t)Nn * Cc * 4);
  size_t oH2bf = take((size_t)Nn * Cc * 2);
  size_t oScore= take((size_t)Ee * 4);
  size_t oOrder= take((size_t)Ee * 4);
  size_t oMC   = take((size_t)Nn * 4);
  size_t oCC   = take((size_t)Nn * 4);
  size_t oMD   = take((size_t)Nn * 4);
  size_t oCD   = take((size_t)Nn * 4);
  size_t oBs   = take(256 * 4);
  size_t oBo   = take(256 * 4);
  size_t oW1ip = take((size_t)FIN * Cc * 2);
  size_t oW1rp = take((size_t)FIN * Cc * 2);
  size_t oW2ip = take((size_t)Cc * Cc * 2);
  size_t oW2rp = take((size_t)Cc * Cc * 2);
  size_t oWm1p = take((size_t)256 * 512 * 2);

  float* deg   = (float*)(ws + oDeg);
  float* dinv  = (float*)(ws + oDinv);
  float* nrm   = (float*)(ws + oNorm);
  unsigned short* xbf  = (unsigned short*)(ws + oXbf);
  float* Hi    = (float*)(ws + oHi);
  float* Hr    = (float*)(ws + oHr);
  float* agg   = (float*)(ws + oAgg);
  unsigned short* h1bf = (unsigned short*)(ws + oH1bf);
  float* h2f   = (float*)(ws + oH2f);
  unsigned short* h2bf = (unsigned short*)(ws + oH2bf);
  float* score = (float*)(ws + oScore);
  int*   order = (int*)(ws + oOrder);
  int*   maskC = (int*)(ws + oMC);
  int*   cumC  = (int*)(ws + oCC);
  int*   maskD = (int*)(ws + oMD);
  int*   cumD  = (int*)(ws + oCD);
  int*   bsum  = (int*)(ws + oBs);
  int*   boff  = (int*)(ws + oBo);
  unsigned short* w1ip = (unsigned short*)(ws + oW1ip);
  unsigned short* w1rp = (unsigned short*)(ws + oW1rp);
  unsigned short* w2ip = (unsigned short*)(ws + oW2ip);
  unsigned short* w2rp = (unsigned short*)(ws + oW2rp);
  unsigned short* wm1p = (unsigned short*)(ws + oWm1p);

  // ---- output layout (flat floats, reference tuple order) ----
  float* out = (float*)d_out;
  const size_t NC = (size_t)Nn * Cc;
  float* cX   = out;                    // NC
  float* cEi  = cX + NC;                // 2*EK
  float* cAt  = cEi + 2 * (size_t)EK;   // EK
  float* cW   = cAt + EK;               // EK
  float* cB   = cW + EK;                // Nn
  float* cN   = cB + Nn;                // 1
  float* fX   = cN + 1;                 // NC
  float* fEi  = fX + NC;
  float* fAt  = fEi + 2 * (size_t)DK;
  float* fW   = fAt + DK;
  float* fB   = fW + DK;
  float* fN   = fB + Nn;
  float* oScr = fN + 1;                 // Ee

  const int T = 256;
  // --- degree / norm ---
  hipMemsetAsync(deg, 0, (size_t)Nn * 4, stream);
  k_deg <<<(Ee + T - 1) / T, T, 0, stream>>>(ei, attr, deg);
  k_dinv<<<(Nn + T - 1) / T, T, 0, stream>>>(deg, dinv);
  k_norm<<<(Ee + T - 1) / T, T, 0, stream>>>(ei, attr, dinv, nrm);

  // --- convert x, pack weights ---
  k_cvt_bf16<<<(Nn * FIN / 4 + T - 1) / T, T, 0, stream>>>(x, xbf, Nn * FIN / 4);
  k_pack_w<<<(FIN * Cc + T - 1) / T, T, 0, stream>>>(W1i, w1ip, FIN, Cc);
  k_pack_w<<<(FIN * Cc + T - 1) / T, T, 0, stream>>>(W1r, w1rp, FIN, Cc);
  k_pack_w<<<(Cc * Cc + T - 1) / T, T, 0, stream>>>(W2i, w2ip, Cc, Cc);
  k_pack_w<<<(Cc * Cc + T - 1) / T, T, 0, stream>>>(W2r, w2rp, Cc, Cc);
  k_pack_w<<<(256 * 512 + T - 1) / T, T, 0, stream>>>(Wm1, wm1p, 256, 512);

  // --- ARMA layer 1 ---
  k_gemm_bf16<8><<<Nn / 128, T, 0, stream>>>(xbf, w1ip, Hi, FIN, Cc);
  k_gemm_bf16<8><<<Nn / 128, T, 0, stream>>>(xbf, w1rp, Hr, FIN, Cc);
  hipMemsetAsync(agg, 0, (size_t)Nn * Cc * 4, stream);
  k_scatter<<<Ee * 32 / T, T, 0, stream>>>(ei, nrm, Hi, agg);
  k_combine<<<Nn * 32 / T, T, 0, stream>>>(agg, Hr, b1, h1bf, (float*)nullptr);

  // --- ARMA layer 2 ---
  k_gemm_bf16<8><<<Nn / 128, T, 0, stream>>>(h1bf, w2ip, Hi, Cc, Cc);
  k_gemm_bf16<8><<<Nn / 128, T, 0, stream>>>(h1bf, w2rp, Hr, Cc, Cc);
  hipMemsetAsync(agg, 0, (size_t)Nn * Cc * 4, stream);
  k_scatter<<<Ee * 32 / T, T, 0, stream>>>(ei, nrm, Hi, agg);
  k_combine<<<Nn * 32 / T, T, 0, stream>>>(agg, Hr, b2, h2bf, h2f);

  // --- fused edge MLP: gather straight into LDS, no HBM rep buffer ---
  k_mlp_score<<<Ee / MLP_ROWS, 128, 0, stream>>>(ei, h2bf, wm1p, bm1, Wm2, bm2, score);

  // --- per-group top-K via bitonic sort ---
  k_sort_group<<<Bg, 512, 0, stream>>>(score, order);

  // --- relabel: causal ---
  hipMemsetAsync(maskC, 0, (size_t)Nn * 4, stream);
  k_mark<<<(EK + T - 1) / T, T, 0, stream>>>(ei, order, maskC, 0, EK);
  k_scan1<<<Nn / 256, 256, 0, stream>>>(maskC, cumC, bsum);
  k_scan2<<<1, 256, 0, stream>>>(bsum, boff);
  k_scan3<<<Nn / 256, 256, 0, stream>>>(cumC, boff);
  // --- relabel: conf ---
  hipMemsetAsync(maskD, 0, (size_t)Nn * 4, stream);
  k_mark<<<(DK + T - 1) / T, T, 0, stream>>>(ei, order, maskD, Kk, DK);
  k_scan1<<<Nn / 256, 256, 0, stream>>>(maskD, cumD, bsum);
  k_scan2<<<1, 256, 0, stream>>>(bsum, boff);
  k_scan3<<<Nn / 256, 256, 0, stream>>>(cumD, boff);

  // --- outputs ---
  k_edge_out<<<(EK + T - 1) / T, T, 0, stream>>>(ei, order, cumC, attr, score,
                                                 cEi, cAt, cW, 0, EK, 1.f);
  k_edge_out<<<(DK + T - 1) / T, T, 0, stream>>>(ei, order, cumD, attr, score,
                                                 fEi, fAt, fW, Kk, DK, -1.f);
  k_node_x<<<Nn * 32 / T, T, 0, stream>>>(maskC, cumC, h2f, cX);
  k_node_x<<<Nn * 32 / T, T, 0, stream>>>(maskD, cumD, h2f, fX);
  k_node_meta<<<(Nn + T - 1) / T, T, 0, stream>>>(maskC, cumC, bat, cB, cN);
  k_node_meta<<<(Nn + T - 1) / T, T, 0, stream>>>(maskD, cumD, bat, fB, fN);
  k_copy<<<(Ee + T - 1) / T, T, 0, stream>>>(score, oScr, Ee);
}
